// MoeDecoderLayer_63891933495372
// MI455X (gfx1250) — compile-verified, matrix operand order checked by reading, not run on hardware
//
#include <hip/hip_runtime.h>
#include <hip/hip_bf16.h>

// ---- problem constants (match reference) ----
#define BB   2
#define SS   2048
#define HH   2048
#define NHQ  16
#define NKVH 4
#define DHD  128
#define NE   8
#define MID  1408
#define TDIM (BB * SS)   // 4096 tokens

typedef __attribute__((ext_vector_type(16))) __bf16 v16bf;
typedef __attribute__((ext_vector_type(8)))  __bf16 v8bf;
typedef __attribute__((ext_vector_type(8)))  float  v8f;
typedef unsigned int u32x4 __attribute__((ext_vector_type(4)));
typedef int          i32x8 __attribute__((ext_vector_type(8)));
typedef int          i32x4 __attribute__((ext_vector_type(4)));

// TDM (tensor_load_to_lds): this toolchain exposes the 6-arg form
// (uint32x4 g0, int32x8 g1, int32x4 g2, int32x4 g3, int32x8 extra, i32 cpol).
#if defined(__has_builtin)
#  if __has_builtin(__builtin_amdgcn_tensor_load_to_lds) && \
      __has_builtin(__builtin_amdgcn_s_wait_tensorcnt)
#    define MOE_TDM 1
#  else
#    define MOE_TDM 0
#  endif
#else
#  define MOE_TDM 0
#endif
#if MOE_TDM
#  define XA_LD 2048           // TDM writes dense rows
#else
#  define XA_LD 2056           // +16B row pad: kills 16-way LDS bank conflicts
#endif

union FragU { v16bf v; v8bf h[2]; };

// Load one 16x32 bf16 WMMA A/B fragment from a row-major [16, >=32] tile.
// Per ISA 7.12.2: lane L holds row L%16; lanes 0-15 K={0..7,16..23},
// lanes 16-31 K={8..15,24..31}  -> two contiguous 16B loads per lane.
__device__ __forceinline__ v16bf frag_from(const __bf16* p) {
  FragU u;
  u.h[0] = *(const v8bf*)p;
  u.h[1] = *(const v8bf*)(p + 16);
  return u.v;
}
__device__ __forceinline__ v16bf load_frag(const __bf16* base, int ld) {
  int lane = threadIdx.x & 31;
  return frag_from(base + (size_t)(lane & 15) * ld + ((lane & 16) ? 8 : 0));
}
__device__ __forceinline__ v8f wmma_bf16(v16bf a, v16bf b, v8f c) {
  return __builtin_amdgcn_wmma_f32_16x16x32_bf16(false, a, false, b, (short)0, c,
                                                 false, false);
}

// ---- fp32 -> bf16 transpose (weights become [N,K] so B-frag loads == A-frag loads)
__global__ __launch_bounds__(256) void transpose_bf16_kernel(
    const float* __restrict__ src, __bf16* __restrict__ dst, int R, int C) {
  size_t nn = (size_t)R * C;
  const float* s = src + (size_t)blockIdx.z * nn;
  __bf16* d = dst + (size_t)blockIdx.z * nn;
  int idx = blockIdx.x * 256 + threadIdx.x;
  if (idx < R * C) {
    int r = idx / C, c = idx % C;
    d[(size_t)c * R + r] = (__bf16)s[idx];
  }
}

// ---- RMSNorm (fp32 in, bf16 out), one block per token row ----
__global__ __launch_bounds__(256) void rmsnorm_kernel(
    const float* __restrict__ x, const float* __restrict__ w,
    __bf16* __restrict__ outp) {
  __shared__ float red[8];
  size_t t = blockIdx.x;
  const float* row = x + t * HH;
  float ss = 0.0f;
  for (int i = threadIdx.x; i < HH; i += 256) { float v = row[i]; ss += v * v; }
#pragma unroll
  for (int off = 16; off; off >>= 1) ss += __shfl_xor(ss, off, 32);
  if ((threadIdx.x & 31) == 0) red[threadIdx.x >> 5] = ss;
  __syncthreads();
  if (threadIdx.x < 8) {
    float v = red[threadIdx.x];
#pragma unroll
    for (int off = 4; off; off >>= 1) v += __shfl_xor(v, off, 32);
    if (threadIdx.x == 0) red[0] = v;
  }
  __syncthreads();
  float inv = rsqrtf(red[0] / (float)HH + 1e-6f);
  for (int i = threadIdx.x; i < HH; i += 256)
    outp[t * HH + i] = (__bf16)(row[i] * inv * w[i]);
}

// ---- generic bf16 WMMA GEMM: C[M,N] = A[M,K] * Bt[N,K]^T ----
// block = 8 waves, tile 64(M) x 128(N); wave = 16x64.
// 2-deep software pipeline + unroll 2 (buffers alternate: no rotate-movs).
__global__ __launch_bounds__(256) void gemm_bf16_kernel(
    const __bf16* __restrict__ A, const __bf16* __restrict__ Bt,
    int M, int N, int K,
    __bf16* __restrict__ outB, float* __restrict__ outF,
    const float* __restrict__ residual) {
  int wave = threadIdx.x >> 5;
  int lane = threadIdx.x & 31;
  int m0 = blockIdx.y * 64 + 16 * (wave & 3);
  int n0 = blockIdx.x * 128 + 64 * (wave >> 2);
  v8f acc[4] = {};
  const __bf16* aptr = A + (size_t)m0 * K;
  const __bf16* bptr[4];
#pragma unroll
  for (int j = 0; j < 4; ++j) bptr[j] = Bt + (size_t)(n0 + 16 * j) * K;

  v16bf a0 = load_frag(aptr, K);
  v16bf b0[4], a1, b1[4];
#pragma unroll
  for (int j = 0; j < 4; ++j) b0[j] = load_frag(bptr[j], K);
  a1 = load_frag(aptr + 32, K);
#pragma unroll
  for (int j = 0; j < 4; ++j) b1[j] = load_frag(bptr[j] + 32, K);

#pragma unroll 2
  for (int k = 64; k < K; k += 64) {
    // warm caches a few steps ahead (pure global addresses -> global_prefetch_b8)
    __builtin_prefetch(aptr + (size_t)(lane & 15) * K + k + 256, 0, 3);
    __builtin_prefetch(Bt + (size_t)(n0 + lane) * K + k + 256, 0, 3);
#pragma unroll
    for (int j = 0; j < 4; ++j) acc[j] = wmma_bf16(a0, b0[j], acc[j]);
    a0 = load_frag(aptr + k, K);
#pragma unroll
    for (int j = 0; j < 4; ++j) b0[j] = load_frag(bptr[j] + k, K);
#pragma unroll
    for (int j = 0; j < 4; ++j) acc[j] = wmma_bf16(a1, b1[j], acc[j]);
    a1 = load_frag(aptr + k + 32, K);
#pragma unroll
    for (int j = 0; j < 4; ++j) b1[j] = load_frag(bptr[j] + k + 32, K);
  }
#pragma unroll
  for (int j = 0; j < 4; ++j) acc[j] = wmma_bf16(a0, b0[j], acc[j]);
#pragma unroll
  for (int j = 0; j < 4; ++j) acc[j] = wmma_bf16(a1, b1[j], acc[j]);

  int half = lane >> 4, cl = lane & 15;
#pragma unroll
  for (int j = 0; j < 4; ++j)
#pragma unroll
    for (int r = 0; r < 8; ++r) {
      size_t row = m0 + r + 8 * half;
      size_t col = n0 + 16 * j + cl;
      float v = acc[j][r];
      if (residual) v += residual[row * N + col];
      if (outF) outF[row * N + col] = v;
      else      outB[row * N + col] = (__bf16)v;
    }
}

// ---- RoPE (in-layout [b,s,h,d] bf16 -> out [b,h,s,d] bf16); pair (d, d+64) per thread
__global__ __launch_bounds__(256) void rope_kernel(
    const __bf16* __restrict__ in, __bf16* __restrict__ outp,
    const float* __restrict__ cosb, const float* __restrict__ sinb, int nh) {
  int idx = blockIdx.x * 256 + threadIdx.x;
  int total = TDIM * nh * 64;
  if (idx >= total) return;
  int d = idx & 63;
  int h = (idx >> 6) % nh;
  int t = idx / (64 * nh);
  int b = t / SS, s = t % SS;
  size_t ce = ((size_t)b * SS + s) * DHD + d;
  float cv = cosb[ce], sv = sinb[ce];
  size_t ib = (size_t)t * (nh * DHD) + h * DHD + d;
  float x0 = (float)in[ib];
  float x1 = (float)in[ib + 64];
  size_t ob = (((size_t)b * nh + h) * SS + s) * DHD + d;
  outp[ob]      = (__bf16)(x0 * cv - x1 * sv);
  outp[ob + 64] = (__bf16)(x1 * cv + x0 * sv);
}

// ---- V: [b,s,kv,d] -> V^T [b,kv,d,s]  (B-operand layout for PV matmul) ----
__global__ __launch_bounds__(256) void vtrans_kernel(
    const __bf16* __restrict__ in, __bf16* __restrict__ outp) {
  int idx = blockIdx.x * 256 + threadIdx.x;
  int total = TDIM * NKVH * DHD;
  if (idx >= total) return;
  int d = idx & 127;
  int h = (idx >> 7) % NKVH;
  int t = idx / (DHD * NKVH);
  int b = t / SS, s = t % SS;
  outp[(((size_t)b * NKVH + h) * DHD + d) * SS + s] =
      in[(size_t)t * (NKVH * DHD) + h * DHD + d];
}

// ---- flash attention: 1 wave = 16 q rows, online softmax, K-steps of 32 ----
__global__ __launch_bounds__(128) void attn_kernel(
    const __bf16* __restrict__ Q, const __bf16* __restrict__ Kb,
    const __bf16* __restrict__ Vt, __bf16* __restrict__ O) {
  __shared__ __bf16 pbuf[4][16 * 32];   // per-wave P staging (C-layout -> A-frag)
  int wave = threadIdx.x >> 5;
  int lane = threadIdx.x & 31;
  int bh = blockIdx.x;
  int b = bh / NHQ, h = bh % NHQ;
  int kv = h / (NHQ / NKVH);
  int q0 = blockIdx.y * 64 + 16 * wave;
  const __bf16* qbase = Q + ((size_t)bh * SS + q0) * DHD;
  const __bf16* kbase = Kb + (size_t)(b * NKVH + kv) * SS * DHD;
  const __bf16* vbase = Vt + (size_t)(b * NKVH + kv) * DHD * SS;
  __bf16* myp = pbuf[wave];

  v16bf aq[4];
#pragma unroll
  for (int kk = 0; kk < 4; ++kk) aq[kk] = load_frag(qbase + 32 * kk, DHD);

  v8f o[8] = {};
  float m[8], l[8];
#pragma unroll
  for (int r = 0; r < 8; ++r) { m[r] = -1e30f; l[r] = 0.0f; }
  int half = lane >> 4, cl = lane & 15;
  const float scale = 0.08838834764831845f;  // 1/sqrt(128)

  for (int ks = 0; ks < SS; ks += 32) {
    // load all 8 K fragments of this step up-front: one wait, 8 QK WMMAs
    v16bf kf[8];
#pragma unroll
    for (int i = 0; i < 8; ++i)
      kf[i] = load_frag(kbase + (size_t)(ks + (i >= 4 ? 16 : 0)) * DHD + 32 * (i & 3), DHD);
    if (ks + 32 < SS) {
      // warm next K block + V block while softmax VALU work runs
      __builtin_prefetch(kbase + (size_t)(ks + 32 + lane) * DHD, 0, 3);
      __builtin_prefetch(vbase + (size_t)(lane * 4) * SS + ks + 32, 0, 3);
    }
    v8f s0 = {}, s1 = {};
#pragma unroll
    for (int kk = 0; kk < 4; ++kk) s0 = wmma_bf16(aq[kk], kf[kk], s0);
#pragma unroll
    for (int kk = 0; kk < 4; ++kk) s1 = wmma_bf16(aq[kk], kf[4 + kk], s1);
#pragma unroll
    for (int r = 0; r < 8; ++r) {
      float x0 = s0[r] * scale, x1 = s1[r] * scale;
      float mx = fmaxf(x0, x1);
#pragma unroll
      for (int off = 1; off < 16; off <<= 1) mx = fmaxf(mx, __shfl_xor(mx, off, 32));
      float mnew = fmaxf(m[r], mx);
      float sc = __expf(m[r] - mnew);
      float p0 = __expf(x0 - mnew), p1 = __expf(x1 - mnew);
      float rs = p0 + p1;
#pragma unroll
      for (int off = 1; off < 16; off <<= 1) rs += __shfl_xor(rs, off, 32);
      l[r] = l[r] * sc + rs;
      m[r] = mnew;
#pragma unroll
      for (int j = 0; j < 8; ++j) o[j][r] *= sc;
      int row = r + 8 * half;
      myp[row * 32 + cl]      = (__bf16)p0;
      myp[row * 32 + cl + 16] = (__bf16)p1;
    }
    v16bf pa = load_frag(myp, 32);   // same-wave LDS dep; compiler inserts s_wait_dscnt
#pragma unroll
    for (int j = 0; j < 8; ++j)
      o[j] = wmma_bf16(pa, load_frag(vbase + (size_t)(j * 16) * SS + ks, SS), o[j]);
  }
#pragma unroll
  for (int j = 0; j < 8; ++j)
#pragma unroll
    for (int r = 0; r < 8; ++r) {
      int srow = q0 + r + 8 * half;
      O[((size_t)b * SS + srow) * (NHQ * DHD) + h * DHD + j * 16 + cl] =
          (__bf16)(o[j][r] / l[r]);
    }
}

// ---- router: softmax(top-2) + build per-expert token/weight lists ----
__global__ __launch_bounds__(256) void router_kernel(
    const __bf16* __restrict__ X, const float* __restrict__ Wr,
    int* __restrict__ cnt, int* __restrict__ tlist, float* __restrict__ wlist) {
  int t = blockIdx.x * 8 + (threadIdx.x >> 5);
  int lane = threadIdx.x & 31;
  if (t >= TDIM) return;
  float acc[NE] = {};
  const __bf16* xr = X + (size_t)t * HH;
  for (int hi = lane; hi < HH; hi += 32) {
    float xv = (float)xr[hi];
#pragma unroll
    for (int e = 0; e < NE; ++e) acc[e] += xv * Wr[(size_t)hi * NE + e];
  }
#pragma unroll
  for (int e = 0; e < NE; ++e)
#pragma unroll
    for (int off = 16; off; off >>= 1) acc[e] += __shfl_xor(acc[e], off, 32);
  if (lane == 0) {
    float mx = acc[0];
    for (int e = 1; e < NE; ++e) mx = fmaxf(mx, acc[e]);
    float p[NE], den = 0.0f;
    for (int e = 0; e < NE; ++e) { p[e] = __expf(acc[e] - mx); den += p[e]; }
    for (int e = 0; e < NE; ++e) p[e] /= den;
    int i1 = 0;
    for (int e = 1; e < NE; ++e) if (p[e] > p[i1]) i1 = e;
    int i2 = (i1 == 0) ? 1 : 0;
    for (int e = 0; e < NE; ++e) if (e != i1 && p[e] > p[i2]) i2 = e;
    float s = p[i1] + p[i2];
    int q1 = atomicAdd(&cnt[i1], 1);
    tlist[i1 * TDIM + q1] = t; wlist[i1 * TDIM + q1] = p[i1] / s;
    int q2 = atomicAdd(&cnt[i2], 1);
    tlist[i2 * TDIM + q2] = t; wlist[i2 * TDIM + q2] = p[i2] / s;
  }
}

// ---- fused routed expert FFN ----
// A-tile (16 gathered tokens x 2048 bf16, 64KB) staged in LDS ONCE:
//   - TDM gather mode (tensor_load_to_lds + s_wait_tensorcnt) when available
//   - cooperative vector copy fallback (with bank-conflict row padding)
// phase 1: h1 = silu(x@gate)*(x@up) into LDS; phase 2: down-proj + scatter.
__global__ __launch_bounds__(256) void moe_kernel(
    const __bf16* __restrict__ X, const __bf16* __restrict__ gT,
    const __bf16* __restrict__ uT, const __bf16* __restrict__ dT,
    const int* __restrict__ cnt, const int* __restrict__ tlist,
    const float* __restrict__ wlist, float* __restrict__ out) {
  __shared__ __bf16 xa[16 * XA_LD];  // gathered activation tile (64KB)
  __shared__ __bf16 h1s[16 * MID];   // 45KB: 16 tokens x 1408 hidden
  int e = blockIdx.x;
  int c = cnt[e];
  int m0 = blockIdx.y * 16;
  if (m0 >= c) return;
  int wave = threadIdx.x >> 5;
  int lane = threadIdx.x & 31;
  int half = lane >> 4, cl = lane & 15;
  const __bf16* gbase = gT + (size_t)e * MID * HH;
  const __bf16* ubase = uT + (size_t)e * MID * HH;
  const __bf16* dbase = dT + (size_t)e * HH * MID;

  // uniform gathered token ids for this block (tail rows clamped)
  int gidx[16];
#pragma unroll
  for (int i = 0; i < 16; ++i) {
    int r = m0 + i; if (r >= c) r = c - 1;
    gidx[i] = tlist[e * TDIM + r];
  }

#if MOE_TDM
  if (threadIdx.x < 32) {   // wave 0 drives the Tensor Data Mover
    unsigned int ldsOff = (unsigned int)(uintptr_t)(void*)xa;  // low 32b = LDS byte addr
    unsigned long long ga = (unsigned long long)(uintptr_t)X;
    u32x4 d0; i32x8 d1; i32x4 d2, d3;
    // D# group0: count=1, gather_mode=1, 16-bit indices, lds_addr, 57b global addr, type=2
    d0[0] = 1u | (1u << 31);
    d0[1] = ldsOff;
    d0[2] = (unsigned int)(ga & 0xFFFFFFFFu);
    d0[3] = (unsigned int)((ga >> 32) & 0x01FFFFFFu) | (2u << 30);
    // D# group1: data_size=2B; tensor 2048 x 4096, stride 2048; tile 2048 x 16 indices
    d1[0] = (1 << 16);
    d1[1] = (int)(((unsigned)HH & 0xFFFFu) << 16);
    d1[2] = (int)((((unsigned)HH >> 16) & 0xFFFFu) | (((unsigned)TDIM & 0xFFFFu) << 16));
    d1[3] = (int)((((unsigned)TDIM >> 16) & 0xFFFFu) | ((unsigned)HH << 16));
    d1[4] = 16;        // tile_dim1 = #valid row indices
    d1[5] = HH;        // tensor_dim0_stride (data_size units)
    d1[6] = 0;
    d1[7] = 0;
    // D# groups 2-3: sixteen 16-bit row indices
    d2[0] = gidx[0]  | (gidx[1]  << 16);
    d2[1] = gidx[2]  | (gidx[3]  << 16);
    d2[2] = gidx[4]  | (gidx[5]  << 16);
    d2[3] = gidx[6]  | (gidx[7]  << 16);
    d3[0] = gidx[8]  | (gidx[9]  << 16);
    d3[1] = gidx[10] | (gidx[11] << 16);
    d3[2] = gidx[12] | (gidx[13] << 16);
    d3[3] = gidx[14] | (gidx[15] << 16);
    i32x8 dx = {};     // extra descriptor group (6-arg toolchain form): zero-filled
    __builtin_amdgcn_tensor_load_to_lds(d0, d1, d2, d3, dx, 0);
    __builtin_amdgcn_s_wait_tensorcnt(0);
  }
  __syncthreads();
#else
  // cooperative gather: 16 rows x 2048 bf16, 16B vectors per thread
  for (int i = threadIdx.x; i < 16 * (HH / 8); i += 256) {
    int row = i >> 8;          // HH/8 == 256 segments per row
    int seg = i & 255;
    *(v8bf*)(xa + (size_t)row * XA_LD + seg * 8) =
        *(const v8bf*)(X + (size_t)gidx[row] * HH + seg * 8);
  }
  __syncthreads();
#endif

  // phase 1: h1 = silu(x@gate) * (x@up); wave owns 16 cols per 128-col chunk
  for (int chunk = 0; chunk < MID / 128; ++chunk) {
    int col0 = chunk * 128 + 16 * wave;
    const __bf16* gp = gbase + (size_t)col0 * HH;
    const __bf16* up = ubase + (size_t)col0 * HH;
    v8f ag = {}, au = {};
    v16bf a0 = load_frag(xa, XA_LD);
    v16bf g0 = load_frag(gp, HH);
    v16bf u0 = load_frag(up, HH);
    v16bf a1 = load_frag(xa + 32, XA_LD);
    v16bf g1 = load_frag(gp + 32, HH);
    v16bf u1 = load_frag(up + 32, HH);
#pragma unroll 2
    for (int k = 64; k < HH; k += 64) {
      ag = wmma_bf16(a0, g0, ag);
      au = wmma_bf16(a0, u0, au);
      a0 = load_frag(xa + k, XA_LD);
      g0 = load_frag(gp + k, HH);
      u0 = load_frag(up + k, HH);
      ag = wmma_bf16(a1, g1, ag);
      au = wmma_bf16(a1, u1, au);
      a1 = load_frag(xa + k + 32, XA_LD);
      g1 = load_frag(gp + k + 32, HH);
      u1 = load_frag(up + k + 32, HH);
    }
    ag = wmma_bf16(a0, g0, ag);
    au = wmma_bf16(a0, u0, au);
    ag = wmma_bf16(a1, g1, ag);
    au = wmma_bf16(a1, u1, au);
#pragma unroll
    for (int r = 0; r < 8; ++r) {
      float g = ag[r], u = au[r];
      h1s[(r + 8 * half) * MID + col0 + cl] = (__bf16)((g / (1.0f + __expf(-g))) * u);
    }
  }
  __syncthreads();

  // phase 2: out[t] += w_t * h1 @ down ; wave owns 256 output cols
#pragma unroll 1
  for (int grp = 0; grp < 4; ++grp) {
    int n0 = wave * 256 + grp * 64;
    const __bf16* dp[4];
#pragma unroll
    for (int j = 0; j < 4; ++j) dp[j] = dbase + (size_t)(n0 + 16 * j) * MID;
    v8f acc[4] = {};
    v16bf b0[4], b1[4];
#pragma unroll
    for (int j = 0; j < 4; ++j) b0[j] = load_frag(dp[j], MID);
#pragma unroll
    for (int j = 0; j < 4; ++j) b1[j] = load_frag(dp[j] + 32, MID);
#pragma unroll 2
    for (int k = 64; k < MID; k += 64) {
      v16bf aA = load_frag(h1s + k - 64, MID);
#pragma unroll
      for (int j = 0; j < 4; ++j) acc[j] = wmma_bf16(aA, b0[j], acc[j]);
#pragma unroll
      for (int j = 0; j < 4; ++j) b0[j] = load_frag(dp[j] + k, MID);
      v16bf aB = load_frag(h1s + k - 32, MID);
#pragma unroll
      for (int j = 0; j < 4; ++j) acc[j] = wmma_bf16(aB, b1[j], acc[j]);
#pragma unroll
      for (int j = 0; j < 4; ++j) b1[j] = load_frag(dp[j] + k + 32, MID);
    }
    v16bf aA = load_frag(h1s + MID - 64, MID);
#pragma unroll
    for (int j = 0; j < 4; ++j) acc[j] = wmma_bf16(aA, b0[j], acc[j]);
    v16bf aB = load_frag(h1s + MID - 32, MID);
#pragma unroll
    for (int j = 0; j < 4; ++j) acc[j] = wmma_bf16(aB, b1[j], acc[j]);
#pragma unroll
    for (int j = 0; j < 4; ++j)
#pragma unroll
      for (int r = 0; r < 8; ++r) {
        int gm = m0 + r + 8 * half;
        if (gm < c) {
          int t = tlist[e * TDIM + gm];
          float wgt = wlist[e * TDIM + gm];
          unsafeAtomicAdd(&out[(size_t)t * HH + n0 + 16 * j + cl], wgt * acc[j][r]);
        }
      }
  }
}

extern "C" void kernel_launch(void* const* d_in, const int* in_sizes, int n_in,
                              void* d_out, int out_size, void* d_ws, size_t ws_size,
                              hipStream_t stream) {
  const float* hs   = (const float*)d_in[0];
  const float* cosb = (const float*)d_in[1];
  const float* sinb = (const float*)d_in[2];
  const float* wq   = (const float*)d_in[3];
  const float* wk   = (const float*)d_in[4];
  const float* wv   = (const float*)d_in[5];
  const float* wo   = (const float*)d_in[6];
  const float* ln1  = (const float*)d_in[7];
  const float* ln2  = (const float*)d_in[8];
  const float* wr   = (const float*)d_in[9];
  const float* gw   = (const float*)d_in[10];
  const float* uw   = (const float*)d_in[11];
  const float* dw   = (const float*)d_in[12];
  float* out = (float*)d_out;

  size_t off = 0;
  auto alloc = [&](size_t bytes) -> void* {
    void* p = (char*)d_ws + off;
    off += (bytes + 255) & ~(size_t)255;
    return p;
  };
  __bf16* wqT  = (__bf16*)alloc((size_t)2048 * HH * 2);
  __bf16* wkT  = (__bf16*)alloc((size_t)1024 * HH * 2);
  __bf16* wvT  = (__bf16*)alloc((size_t)1024 * HH * 2);
  __bf16* woT  = (__bf16*)alloc((size_t)HH * 2048 * 2);
  __bf16* gT   = (__bf16*)alloc((size_t)NE * MID * HH * 2);
  __bf16* uT   = (__bf16*)alloc((size_t)NE * MID * HH * 2);
  __bf16* dT   = (__bf16*)alloc((size_t)NE * HH * MID * 2);
  __bf16* xb1  = (__bf16*)alloc((size_t)TDIM * HH * 2);
  __bf16* qtmp = (__bf16*)alloc((size_t)TDIM * 2048 * 2);
  __bf16* ktmp = (__bf16*)alloc((size_t)TDIM * 1024 * 2);
  __bf16* vtmp = (__bf16*)alloc((size_t)TDIM * 1024 * 2);
  __bf16* qb   = (__bf16*)alloc((size_t)TDIM * 2048 * 2);
  __bf16* kb   = (__bf16*)alloc((size_t)TDIM * 1024 * 2);
  __bf16* vTb  = (__bf16*)alloc((size_t)TDIM * 1024 * 2);
  __bf16* aout = (__bf16*)alloc((size_t)TDIM * 2048 * 2);
  float*  hid  = (float*) alloc((size_t)TDIM * HH * 4);
  __bf16* xb2  = (__bf16*)alloc((size_t)TDIM * HH * 2);
  int*    cnt  = (int*)   alloc(NE * sizeof(int));
  int*    tls  = (int*)   alloc((size_t)NE * TDIM * sizeof(int));
  float*  wls  = (float*) alloc((size_t)NE * TDIM * sizeof(float));
  (void)ws_size; (void)in_sizes; (void)n_in; (void)out_size;

  // 1. weights -> bf16, transposed to [N,K]
  transpose_bf16_kernel<<<dim3((HH * 2048 + 255) / 256, 1, 1), 256, 0, stream>>>(wq, wqT, HH, 2048);
  transpose_bf16_kernel<<<dim3((HH * 1024 + 255) / 256, 1, 1), 256, 0, stream>>>(wk, wkT, HH, 1024);
  transpose_bf16_kernel<<<dim3((HH * 1024 + 255) / 256, 1, 1), 256, 0, stream>>>(wv, wvT, HH, 1024);
  transpose_bf16_kernel<<<dim3((2048 * HH + 255) / 256, 1, 1), 256, 0, stream>>>(wo, woT, 2048, HH);
  transpose_bf16_kernel<<<dim3((HH * MID + 255) / 256, 1, NE), 256, 0, stream>>>(gw, gT, HH, MID);
  transpose_bf16_kernel<<<dim3((HH * MID + 255) / 256, 1, NE), 256, 0, stream>>>(uw, uT, HH, MID);
  transpose_bf16_kernel<<<dim3((MID * HH + 255) / 256, 1, NE), 256, 0, stream>>>(dw, dT, MID, HH);

  // 2. RMSNorm 1
  rmsnorm_kernel<<<TDIM, 256, 0, stream>>>(hs, ln1, xb1);

  // 3. QKV projections (bf16 WMMA)
  gemm_bf16_kernel<<<dim3(2048 / 128, TDIM / 64), 256, 0, stream>>>(xb1, wqT, TDIM, 2048, HH, qtmp, nullptr, nullptr);
  gemm_bf16_kernel<<<dim3(1024 / 128, TDIM / 64), 256, 0, stream>>>(xb1, wkT, TDIM, 1024, HH, ktmp, nullptr, nullptr);
  gemm_bf16_kernel<<<dim3(1024 / 128, TDIM / 64), 256, 0, stream>>>(xb1, wvT, TDIM, 1024, HH, vtmp, nullptr, nullptr);

  // 4. RoPE + layout changes
  rope_kernel<<<(TDIM * NHQ * 64 + 255) / 256, 256, 0, stream>>>(qtmp, qb, cosb, sinb, NHQ);
  rope_kernel<<<(TDIM * NKVH * 64 + 255) / 256, 256, 0, stream>>>(ktmp, kb, cosb, sinb, NKVH);
  vtrans_kernel<<<(TDIM * NKVH * DHD + 255) / 256, 256, 0, stream>>>(vtmp, vTb);

  // 5. attention
  attn_kernel<<<dim3(BB * NHQ, SS / 64), 128, 0, stream>>>(qb, kb, vTb, aout);

  // 6. O-proj + residual -> hidden (f32)
  gemm_bf16_kernel<<<dim3(HH / 128, TDIM / 64), 256, 0, stream>>>(aout, woT, TDIM, HH, 2048, nullptr, hid, hs);

  // 7. RMSNorm 2
  rmsnorm_kernel<<<TDIM, 256, 0, stream>>>(hid, ln2, xb2);

  // 8. router + routing lists
  (void)hipMemsetAsync(cnt, 0, NE * sizeof(int), stream);
  router_kernel<<<(TDIM + 7) / 8, 256, 0, stream>>>(xb2, wr, cnt, tls, wls);

  // 9. d_out = hidden (residual), then MoE scatters weighted expert output
  (void)hipMemcpyAsync(out, hid, (size_t)TDIM * HH * 4, hipMemcpyDeviceToDevice, stream);
  moe_kernel<<<dim3(NE, TDIM / 16), 256, 0, stream>>>(xb2, gT, uT, dT, cnt, tls, wls, out);
}
